// ScaleAwareLogRatioConditionalGraphNetwork_49830210568749
// MI455X (gfx1250) — compile-verified, matrix operand order checked by reading, not run on hardware
//
#include <hip/hip_runtime.h>
#include <hip/hip_bf16.h>

// ---------------------------------------------------------------------------
// ScaleAwareLogRatioConditionalGraphNetwork for MI455X (gfx1250, wave32).
// Hidden states stored f16, all H=64 GEMMs run on v_wmma_f32_16x16x32_f16
// (f32 accumulate). 64 rows per block (4 row-tiles x 4 waves): each B (weight)
// fragment is loaded once and reused across 4 A tiles -> 4x less L2 weight
// traffic, 4 WMMAs per B-fragment load.
// ---------------------------------------------------------------------------

typedef __attribute__((ext_vector_type(16))) _Float16 v16h;
typedef __attribute__((ext_vector_type(8)))  _Float16 v8h;
typedef __attribute__((ext_vector_type(8)))  float    v8f;

#define HDIM 64
#define RT   4      // row tiles per block (64 rows)

// --- WMMA fragment load: 16 rows x 32 K from row-major f16, leading dim ld.
// 16-bit A layout (ISA 7.12.2): lane l (m = l&15, g = l>>4):
//   halves 0..7  -> K = kc + g*8 + 0..7
//   halves 8..15 -> K = kc + 16 + g*8 + 0..7
// B fragment uses the identical pattern on W^T rows (n = l&15).
__device__ __forceinline__ v16h load_frag_rm(const _Float16* M, int ld, int kc, int lane) {
  int r = lane & 15;
  int g = lane >> 4;
  const _Float16* p = M + (size_t)r * ld + kc + g * 8;
  v8h lo = *(const v8h*)p;
  v8h hi = *(const v8h*)(p + 16);
  v16h f;
#pragma unroll
  for (int i = 0; i < 8; ++i) { f[i] = lo[i]; f[i + 8] = hi[i]; }
  return f;
}

// NT row-tiles x 16 cols: IN(NT*16 x K) * W(Kx64)[cols of this wave].
// wt_col points at W^T row block for this wave's 16 columns ([16][K] f16).
// B fragment loaded once per kc chunk, reused for all NT A tiles.
template <int K, int NT>
__device__ __forceinline__ void gemm_rows(const _Float16* in_tile, const _Float16* wt_col,
                                          v8f acc[NT], int lane) {
#pragma unroll
  for (int kc = 0; kc < K; kc += 32) {
    v16h b = load_frag_rm(wt_col, K, kc, lane);
#pragma unroll
    for (int t = 0; t < NT; ++t) {
      v16h a = load_frag_rm(in_tile + (size_t)t * 16 * K, K, kc, lane);
      acc[t] = __builtin_amdgcn_wmma_f32_16x16x32_f16(false, a, false, b, (short)0, acc[t],
                                                      false, false);
    }
  }
}

__device__ __forceinline__ void cp64h(_Float16* d, const _Float16* s) {
  const v8h* sv = (const v8h*)s;
  v8h* dv = (v8h*)d;
#pragma unroll
  for (int i = 0; i < 8; ++i) dv[i] = sv[i];
}

__device__ __forceinline__ int imin(int a, int b) { return a < b ? a : b; }

// --- f32 -> f16 transposed weight pack: wt[j*K+k] = w[k*64+j]
__global__ void k_transpose(const float* __restrict__ w, _Float16* __restrict__ wt, int K) {
  int idx = blockIdx.x * 256 + threadIdx.x;
  if (idx < K * HDIM) {
    int j = idx / K, k = idx - j * K;
    wt[idx] = (_Float16)w[k * HDIM + j];
  }
}

__global__ void k_zero(float* p, size_t n) {
  size_t i = (size_t)blockIdx.x * 256 + threadIdx.x;
  if (i < n) p[i] = 0.f;
}

// --- per-src-node edge counts + per-graph edge counts
__global__ void k_count_edges(const int* __restrict__ src, const int* __restrict__ batch,
                              float* cnt_src, float* eg_cnt, int n_edges) {
  __shared__ float g[16];
  int tid = threadIdx.x;
  if (tid < 16) g[tid] = 0.f;
  __syncthreads();
  int e = blockIdx.x * 256 + tid;
  if (e < n_edges) {
    int s = src[e];
    atomicAdd(&cnt_src[s], 1.f);
    atomicAdd(&g[batch[s]], 1.f);
  }
  __syncthreads();
  if (tid < 16) atomicAdd(&eg_cnt[tid], g[tid]);
}

__global__ void k_count_nodes(const int* __restrict__ batch, float* ng_cnt, int n_nodes) {
  __shared__ float g[16];
  int tid = threadIdx.x;
  if (tid < 16) g[tid] = 0.f;
  __syncthreads();
  int n = blockIdx.x * 256 + tid;
  if (n < n_nodes) atomicAdd(&g[batch[n]], 1.f);
  __syncthreads();
  if (tid < 16) atomicAdd(&ng_cnt[tid], g[tid]);
}

// --- cond_enc / scale_enc / u_enc chain on 16 graphs -> u[16][64] f16. 1 block.
__global__ void k_small_u(const float* cond, const float* scal,
                          const float* cw1, const float* cb1, const float* cw2, const float* cb2,
                          const float* sw1, const float* sb1, const float* sw2, const float* sb2,
                          const float* uw1, const float* ub1, const float* uw2, const float* ub2,
                          _Float16* u) {
  __shared__ float hid[16 * 64];
  __shared__ float cat[16 * 128];
  int c = threadIdx.x;  // 64 threads
  for (int r = 0; r < 16; r++) {
    float v = cb1[c];
    for (int k = 0; k < 8; k++) v += cond[r * 8 + k] * cw1[k * 64 + c];
    hid[r * 64 + c] = fmaxf(v, 0.f);
  }
  __syncthreads();
  for (int r = 0; r < 16; r++) {
    float v = cb2[c];
    for (int k = 0; k < 64; k++) v += hid[r * 64 + k] * cw2[k * 64 + c];
    cat[r * 128 + c] = v;
  }
  __syncthreads();
  for (int r = 0; r < 16; r++) {
    float v = sb1[c];
    for (int k = 0; k < 6; k++) v += scal[r * 6 + k] * sw1[k * 64 + c];
    hid[r * 64 + c] = fmaxf(v, 0.f);
  }
  __syncthreads();
  for (int r = 0; r < 16; r++) {
    float v = sb2[c];
    for (int k = 0; k < 64; k++) v += hid[r * 64 + k] * sw2[k * 64 + c];
    cat[r * 128 + 64 + c] = v;
  }
  __syncthreads();
  for (int r = 0; r < 16; r++) {
    float v = ub1[c];
    for (int k = 0; k < 128; k++) v += cat[r * 128 + k] * uw1[k * 64 + c];
    hid[r * 64 + c] = fmaxf(v, 0.f);
  }
  __syncthreads();
  for (int r = 0; r < 16; r++) {
    float v = ub2[c];
    for (int k = 0; k < 64; k++) v += hid[r * 64 + k] * uw2[k * 64 + c];
    u[r * 64 + c] = (_Float16)v;
  }
}

// --- encoders: DIN->64 (VALU, relu) then 64->64 (WMMA). 64 rows/block, 128 thr.
template <int DIN>
__global__ void k_enc(const float* __restrict__ in, const float* __restrict__ w1,
                      const float* __restrict__ b1, const _Float16* __restrict__ wt2,
                      const float* __restrict__ b2, _Float16* __restrict__ out, int n_rows) {
  __shared__ __align__(16) _Float16 hid[64 * 64];
  int tile = blockIdx.x * 64;
  int tid = threadIdx.x, lane = tid & 31, wave = tid >> 5;

  {  // first layer: 2 threads per row, 32 cols each
    int row = tid >> 1;
    int c0 = (tid & 1) * 32;
    int n = imin(tile + row, n_rows - 1);
    float xin[DIN];
#pragma unroll
    for (int k = 0; k < DIN; k++) xin[k] = in[(size_t)n * DIN + k];
#pragma unroll
    for (int c = 0; c < 32; c++) {
      float v = b1[c0 + c];
#pragma unroll
      for (int k = 0; k < DIN; k++) v += xin[k] * w1[k * 64 + c0 + c];
      hid[row * 64 + c0 + c] = (_Float16)fmaxf(v, 0.f);
    }
  }
  __syncthreads();

  v8f acc[RT] = {};
  gemm_rows<64, RT>(hid, wt2 + (size_t)wave * 16 * 64, acc, lane);
  int col = wave * 16 + (lane & 15);
#pragma unroll
  for (int t = 0; t < RT; ++t)
#pragma unroll
    for (int r = 0; r < 8; ++r) {
      int rr = t * 16 + r + 8 * (lane >> 4);
      int nn = tile + rr;
      if (nn < n_rows) out[(size_t)nn * 64 + col] = (_Float16)(acc[t][r] + b2[col]);
    }
}

// --- edge update: eh[e] = MLP256([xh[src]|xh[dst]|eh|u_edge]) in place.
__global__ void k_edge_layer(const _Float16* __restrict__ xh, _Float16* __restrict__ eh,
                             const _Float16* __restrict__ u, const int* __restrict__ src,
                             const int* __restrict__ dst, const int* __restrict__ batch,
                             const _Float16* __restrict__ wt1, const float* __restrict__ b1,
                             const _Float16* __restrict__ wt2, const float* __restrict__ b2,
                             int n_edges) {
  __shared__ __align__(16) _Float16 in_lds[64 * 256];  // 32 KB
  __shared__ __align__(16) _Float16 hid[64 * 64];      //  8 KB
  int tile = blockIdx.x * 64;
  int tid = threadIdx.x, lane = tid & 31, wave = tid >> 5;

  // gather: 256 jobs (64 rows x 4 sections), 64 halves each
  for (int job = tid; job < 256; job += 128) {
    int row = job >> 2, sec = job & 3;
    int e = imin(tile + row, n_edges - 1);
    const _Float16* sp;
    if (sec == 0)      sp = xh + (size_t)src[e] * 64;
    else if (sec == 1) sp = xh + (size_t)dst[e] * 64;
    else if (sec == 2) sp = eh + (size_t)e * 64;
    else               sp = u + (size_t)batch[src[e]] * 64;
    cp64h(&in_lds[row * 256 + sec * 64], sp);
  }
  __syncthreads();

  int col = wave * 16 + (lane & 15);
  v8f acc[RT] = {};
  gemm_rows<256, RT>(in_lds, wt1 + (size_t)wave * 16 * 256, acc, lane);
#pragma unroll
  for (int t = 0; t < RT; ++t)
#pragma unroll
    for (int r = 0; r < 8; ++r) {
      int rr = t * 16 + r + 8 * (lane >> 4);
      hid[rr * 64 + col] = (_Float16)fmaxf(acc[t][r] + b1[col], 0.f);
    }
  __syncthreads();

  v8f acc2[RT] = {};
  gemm_rows<64, RT>(hid, wt2 + (size_t)wave * 16 * 64, acc2, lane);
#pragma unroll
  for (int t = 0; t < RT; ++t)
#pragma unroll
    for (int r = 0; r < 8; ++r) {
      int rr = t * 16 + r + 8 * (lane >> 4);
      int e = tile + rr;
      if (e < n_edges) eh[(size_t)e * 64 + col] = (_Float16)(acc2[t][r] + b2[col]);
    }
}

// --- scatter edges into per-node sums (agg pre-zeroed)
__global__ void k_scatter_edge(const _Float16* __restrict__ eh, const int* __restrict__ src,
                               float* agg_sum, int n_edges) {
  int tid = threadIdx.x;               // 256 threads, 4 edges/block
  int e = blockIdx.x * 4 + (tid >> 6);
  int c = tid & 63;
  if (e < n_edges)
    atomicAdd(&agg_sum[(size_t)src[e] * 64 + c], (float)eh[(size_t)e * 64 + c]);
}

// --- node update: xh[n] += MLP192([xh|agg_mean|u_node]) in place.
__global__ void k_node_layer(_Float16* __restrict__ xh, const float* __restrict__ agg_sum,
                             const float* __restrict__ cnt_src, const _Float16* __restrict__ u,
                             const int* __restrict__ batch, const _Float16* __restrict__ wt1,
                             const float* __restrict__ b1, const _Float16* __restrict__ wt2,
                             const float* __restrict__ b2, int n_nodes) {
  __shared__ __align__(16) _Float16 in_lds[64 * 192];  // 24 KB
  __shared__ __align__(16) _Float16 hid[64 * 64];      //  8 KB
  int tile = blockIdx.x * 64;
  int tid = threadIdx.x, lane = tid & 31, wave = tid >> 5;

  // gather: 192 jobs (64 rows x 3 sections), 64 halves each
  for (int job = tid; job < 192; job += 128) {
    int row = job / 3, sec = job - row * 3;
    int n = imin(tile + row, n_nodes - 1);
    _Float16* d = &in_lds[row * 192 + sec * 64];
    if (sec == 0) {
      cp64h(d, xh + (size_t)n * 64);
    } else if (sec == 1) {
      float inv = 1.f / fmaxf(cnt_src[n], 1.f);
      const float* s = agg_sum + (size_t)n * 64;
#pragma unroll
      for (int i = 0; i < 64; i++) d[i] = (_Float16)(s[i] * inv);
    } else {
      cp64h(d, u + (size_t)batch[n] * 64);
    }
  }
  __syncthreads();

  int col = wave * 16 + (lane & 15);
  v8f acc[RT] = {};
  gemm_rows<192, RT>(in_lds, wt1 + (size_t)wave * 16 * 192, acc, lane);
#pragma unroll
  for (int t = 0; t < RT; ++t)
#pragma unroll
    for (int r = 0; r < 8; ++r) {
      int rr = t * 16 + r + 8 * (lane >> 4);
      hid[rr * 64 + col] = (_Float16)fmaxf(acc[t][r] + b1[col], 0.f);
    }
  __syncthreads();

  v8f acc2[RT] = {};
  gemm_rows<64, RT>(hid, wt2 + (size_t)wave * 16 * 64, acc2, lane);
#pragma unroll
  for (int t = 0; t < RT; ++t)
#pragma unroll
    for (int r = 0; r < 8; ++r) {
      int rr = t * 16 + r + 8 * (lane >> 4);
      int n = tile + rr;
      if (n < n_nodes) {
        size_t ix = (size_t)n * 64 + col;
        xh[ix] = (_Float16)((float)xh[ix] + acc2[t][r] + b2[col]);
      }
    }
}

// --- node_dec 64->64->64, writes f32 x_out + per-graph sums
__global__ void k_node_dec(const _Float16* __restrict__ xh, const _Float16* __restrict__ wt1,
                           const float* __restrict__ b1, const _Float16* __restrict__ wt2,
                           const float* __restrict__ b2, const int* __restrict__ batch,
                           float* __restrict__ x_out, float* ng_sum, int n_nodes) {
  __shared__ __align__(16) _Float16 hid[64 * 64];
  int tile = blockIdx.x * 64;
  int tid = threadIdx.x, lane = tid & 31, wave = tid >> 5;
  int col = wave * 16 + (lane & 15);

  // A fragments straight from global xh (rows contiguous, ld = 64)
  v8f acc[RT] = {};
  gemm_rows<64, RT>(xh + (size_t)tile * 64, wt1 + (size_t)wave * 16 * 64, acc, lane);
#pragma unroll
  for (int t = 0; t < RT; ++t)
#pragma unroll
    for (int r = 0; r < 8; ++r) {
      int rr = t * 16 + r + 8 * (lane >> 4);
      hid[rr * 64 + col] = (_Float16)fmaxf(acc[t][r] + b1[col], 0.f);
    }
  __syncthreads();

  v8f acc2[RT] = {};
  gemm_rows<64, RT>(hid, wt2 + (size_t)wave * 16 * 64, acc2, lane);
#pragma unroll
  for (int t = 0; t < RT; ++t)
#pragma unroll
    for (int r = 0; r < 8; ++r) {
      int rr = t * 16 + r + 8 * (lane >> 4);
      int n = tile + rr;
      if (n < n_nodes) {
        float v = acc2[t][r] + b2[col];
        x_out[(size_t)n * 64 + col] = v;
        atomicAdd(&ng_sum[batch[n] * 64 + col], v);
      }
    }
}

// --- eh -> per-graph sums, LDS pre-reduction (256 edges/block)
__global__ void k_edge_global(const _Float16* __restrict__ eh, const int* __restrict__ src,
                              const int* __restrict__ batch, float* eg_sum, int n_edges) {
  __shared__ float ls[16 * 64];
  int tid = threadIdx.x;  // 256
  for (int i = tid; i < 1024; i += 256) ls[i] = 0.f;
  __syncthreads();
  int base = blockIdx.x * 256;
  int c = tid & 63;
  for (int ei = tid >> 6; ei < 256; ei += 4) {
    int e = base + ei;
    if (e < n_edges) {
      int b = batch[src[e]];
      atomicAdd(&ls[b * 64 + c], (float)eh[(size_t)e * 64 + c]);
    }
  }
  __syncthreads();
  for (int i = tid; i < 1024; i += 256) atomicAdd(&eg_sum[i], ls[i]);
}

// --- head MLP on the 16 graph embeddings -> log_ratios [16][6]
__global__ void k_head(const float* ng_sum, const float* ng_cnt, const float* eg_sum,
                       const float* eg_cnt, const float* hw1, const float* hb1,
                       const float* hw2, const float* hb2, float* out) {
  __shared__ float cat[16 * 128];
  __shared__ float hid[16 * 64];
  int c = threadIdx.x;  // 64 threads
  for (int r = 0; r < 16; r++) {
    cat[r * 128 + c]      = ng_sum[r * 64 + c] / fmaxf(ng_cnt[r], 1.f);
    cat[r * 128 + 64 + c] = eg_sum[r * 64 + c] / fmaxf(eg_cnt[r], 1.f);
  }
  __syncthreads();
  for (int r = 0; r < 16; r++) {
    float v = hb1[c];
    for (int k = 0; k < 128; k++) v += cat[r * 128 + k] * hw1[k * 64 + c];
    hid[r * 64 + c] = fmaxf(v, 0.f);
  }
  __syncthreads();
  if (c < 6) {
    for (int r = 0; r < 16; r++) {
      float v = hb2[c];
      for (int k = 0; k < 64; k++) v += hid[r * 64 + k] * hw2[k * 6 + c];
      out[r * 6 + c] = v;
    }
  }
}

// ---------------------------------------------------------------------------
struct Mlp { const float *b1, *b2, *w1, *w2; };

extern "C" void kernel_launch(void* const* d_in, const int* in_sizes, int n_in,
                              void* d_out, int out_size, void* d_ws, size_t ws_size,
                              hipStream_t stream) {
  (void)n_in; (void)out_size; (void)ws_size;
  const float* x     = (const float*)d_in[0];
  const int*   eidx  = (const int*)d_in[1];
  const float* eattr = (const float*)d_in[2];
  const float* cond  = (const float*)d_in[3];
  const float* scal  = (const float*)d_in[4];
  const int*   batch = (const int*)d_in[5];
  const int NN = in_sizes[5];
  const int NE = in_sizes[1] / 2;
  const int* src = eidx;
  const int* dst = eidx + NE;

  // params flattened in jax pytree (alphabetical) order, each MLP = b1,b2,w1,w2
  auto M = [&](int base) {
    Mlp m;
    m.b1 = (const float*)d_in[base + 0];
    m.b2 = (const float*)d_in[base + 1];
    m.w1 = (const float*)d_in[base + 2];
    m.w2 = (const float*)d_in[base + 3];
    return m;
  };
  Mlp cond_enc = M(6), edge_enc = M(10), head = M(14);
  Mlp L_edge[3], L_node[3];
  for (int l = 0; l < 3; l++) { L_edge[l] = M(18 + 8 * l); L_node[l] = M(22 + 8 * l); }
  Mlp node_dec = M(42), node_enc = M(46), scale_enc = M(50), u_enc = M(54);

  // workspace carve (256B aligned chunks)
  char* wsp = (char*)d_ws;
  size_t off = 0;
  auto carve = [&](size_t bytes) -> void* {
    void* p = wsp + off;
    off = (off + bytes + 255) & ~(size_t)255;
    return p;
  };
  _Float16* xh = (_Float16*)carve((size_t)NN * 64 * 2);
  _Float16* eh = (_Float16*)carve((size_t)NE * 64 * 2);
  _Float16* u  = (_Float16*)carve(16 * 64 * 2);
  float* agg     = (float*)carve((size_t)NN * 64 * 4);
  float* cnt_src = (float*)carve((size_t)NN * 4);
  float* gbl     = (float*)carve(2080 * 4);
  float* ng_sum = gbl, *ng_cnt = gbl + 1024, *eg_sum = gbl + 1040, *eg_cnt = gbl + 2064;
  auto carveh = [&](int K) { return (_Float16*)carve((size_t)K * 64 * 2); };
  _Float16* wt_nenc2 = carveh(64);
  _Float16* wt_eenc2 = carveh(64);
  _Float16 *wt_e1[3], *wt_e2[3], *wt_n1[3], *wt_n2[3];
  for (int l = 0; l < 3; l++) {
    wt_e1[l] = carveh(256); wt_e2[l] = carveh(64);
    wt_n1[l] = carveh(192); wt_n2[l] = carveh(64);
  }
  _Float16* wt_dec1 = carveh(64);
  _Float16* wt_dec2 = carveh(64);

  // pack weights (f16, transposed)
  auto T = [&](const float* s, _Float16* d, int K) {
    k_transpose<<<(K * 64 + 255) / 256, 256, 0, stream>>>(s, d, K);
  };
  T(node_enc.w2, wt_nenc2, 64);
  T(edge_enc.w2, wt_eenc2, 64);
  for (int l = 0; l < 3; l++) {
    T(L_edge[l].w1, wt_e1[l], 256); T(L_edge[l].w2, wt_e2[l], 64);
    T(L_node[l].w1, wt_n1[l], 192); T(L_node[l].w2, wt_n2[l], 64);
  }
  T(node_dec.w1, wt_dec1, 64);
  T(node_dec.w2, wt_dec2, 64);

  // zero accumulators + counts
  k_zero<<<(2080 + 255) / 256, 256, 0, stream>>>(gbl, 2080);
  k_zero<<<(NN + 255) / 256, 256, 0, stream>>>(cnt_src, (size_t)NN);
  k_count_edges<<<(NE + 255) / 256, 256, 0, stream>>>(src, batch, cnt_src, eg_cnt, NE);
  k_count_nodes<<<(NN + 255) / 256, 256, 0, stream>>>(batch, ng_cnt, NN);

  // global feature u
  k_small_u<<<1, 64, 0, stream>>>(cond, scal,
      cond_enc.w1, cond_enc.b1, cond_enc.w2, cond_enc.b2,
      scale_enc.w1, scale_enc.b1, scale_enc.w2, scale_enc.b2,
      u_enc.w1, u_enc.b1, u_enc.w2, u_enc.b2, u);

  // encoders
  k_enc<6><<<(NN + 63) / 64, 128, 0, stream>>>(x, node_enc.w1, node_enc.b1,
                                               wt_nenc2, node_enc.b2, xh, NN);
  k_enc<4><<<(NE + 63) / 64, 128, 0, stream>>>(eattr, edge_enc.w1, edge_enc.b1,
                                               wt_eenc2, edge_enc.b2, eh, NE);

  // message-passing layers
  for (int l = 0; l < 3; l++) {
    k_edge_layer<<<(NE + 63) / 64, 128, 0, stream>>>(
        xh, eh, u, src, dst, batch,
        wt_e1[l], L_edge[l].b1, wt_e2[l], L_edge[l].b2, NE);
    k_zero<<<(int)(((size_t)NN * 64 + 255) / 256), 256, 0, stream>>>(agg, (size_t)NN * 64);
    k_scatter_edge<<<(NE + 3) / 4, 256, 0, stream>>>(eh, src, agg, NE);
    k_node_layer<<<(NN + 63) / 64, 128, 0, stream>>>(
        xh, agg, cnt_src, u, batch,
        wt_n1[l], L_node[l].b1, wt_n2[l], L_node[l].b2, NN);
  }

  // decode + readout
  float* x_out = (float*)d_out;
  float* lr = x_out + (size_t)NN * 64;
  k_node_dec<<<(NN + 63) / 64, 128, 0, stream>>>(xh, wt_dec1, node_dec.b1, wt_dec2,
                                                 node_dec.b2, batch, x_out, ng_sum, NN);
  k_edge_global<<<(NE + 255) / 256, 256, 0, stream>>>(eh, src, batch, eg_sum, NE);
  k_head<<<1, 64, 0, stream>>>(ng_sum, ng_cnt, eg_sum, eg_cnt,
                               head.w1, head.b1, head.w2, head.b2, lr);
}
